// EdgeAttention_13932873908963
// MI455X (gfx1250) — compile-verified, matrix operand order checked by reading, not run on hardware
//
#include <hip/hip_runtime.h>
#include <hip/hip_bf16.h>
#include <math.h>

// ---------------------------------------------------------------------------
// EdgeAttention (triangle/axial attention) for MI455X (gfx1250, wave32, WMMA)
// All GEMMs run on v_wmma_f32_16x16x32_bf16 with fp32 accumulation.
// Logits GEMM uses double-buffered async global->LDS staging (ASYNCcnt).
// ---------------------------------------------------------------------------

typedef __bf16 v16bf __attribute__((ext_vector_type(16)));
typedef float  v8f   __attribute__((ext_vector_type(8)));
typedef unsigned short ushort8v __attribute__((ext_vector_type(8)));
typedef unsigned int   uint2v   __attribute__((ext_vector_type(2)));
typedef unsigned short u16;

constexpr int L  = 384;
constexpr int DE = 128;          // D_EDGE
constexpr int DB = 64;           // D_BIAS
constexpr int NH = 4;            // heads
constexpr int DH = 48;           // head dim
constexpr int HD = NH * DH;      // 192
constexpr int M  = L * L;        // 147456 rows
constexpr int KD = L * DH;       // 18432 = logits contraction (n,k)
constexpr float INFV = 1000000.0f;

#define DEVFN static __device__ __forceinline__

// --- async global->LDS support (guarded; falls back to sync staging) -------
// Probe round 2 diagnostic showed the builtin signature expects
// (int-vector __device__* src, int-vector lds* dst, imm offset, imm cpol)
// with gcc-style vector_size pointees.
#if defined(__AMDGCN__) && __has_builtin(__builtin_amdgcn_global_load_async_to_lds_b128)
#define HAVE_ASYNC_LDS 1
typedef int i32x4 __attribute__((vector_size(16)));
typedef int i32x2 __attribute__((vector_size(8)));
typedef __attribute__((address_space(1))) i32x4* g_v4p;  // global (== __device__)
typedef __attribute__((address_space(3))) i32x4* l_v4p;  // LDS    (== __shared__)
typedef __attribute__((address_space(1))) i32x2* g_v2p;
typedef __attribute__((address_space(3))) i32x2* l_v2p;
#endif

#if defined(__AMDGCN__) && __has_builtin(__builtin_amdgcn_s_wait_asynccnt)
#define WAIT_ASYNC(n) __builtin_amdgcn_s_wait_asynccnt(n)
#elif defined(__AMDGCN__)
#define WAIT_ASYNC(n) asm volatile("s_wait_asynccnt %0" ::"n"(n))
#else
#define WAIT_ASYNC(n) ((void)0)
#endif

DEVFN u16 f2bf(float f) {
  unsigned u = __float_as_uint(f);
  unsigned r = (u + 0x7FFFu + ((u >> 16) & 1u)) >> 16;  // round-nearest-even
  return (u16)r;
}
DEVFN float bf2f(u16 u) { return __uint_as_float(((unsigned)u) << 16); }
DEVFN __bf16 u2bf(u16 u) { union { u16 s; __bf16 b; } x; x.s = u; return x.b; }

// 16x32 bf16 fragment loader. Per CDNA5 ISA A-layout (CDNA5 ISA):
// lanes 0-15: row M=lane, halves K=[0..7] then K=[16..23];
// lanes 16-31: row M=lane-16, halves K=[8..15] then K=[24..31].
// B operand uses pre-transposed (NxK) storage so the same loader applies.
DEVFN v16bf load_frag(const u16* __restrict__ base, int ld, int lane) {
  const int r  = lane & 15;
  const int kb = (lane >> 4) << 3;  // 0 or 8
  const u16* p = base + (size_t)r * ld + kb;
  ushort8v c0 = *(const ushort8v*)(p);
  ushort8v c1 = *(const ushort8v*)(p + 16);
  v16bf f;
#pragma unroll
  for (int t = 0; t < 8; ++t) { f[t] = u2bf(c0[t]); f[t + 8] = u2bf(c1[t]); }
  return f;
}

DEVFN v8f wmma_acc(v16bf a, v16bf b, v8f c) {
  return __builtin_amdgcn_wmma_f32_16x16x32_bf16(false, a, false, b,
                                                 (short)0, c, false, false);
}

// ---------------------------------------------------------------------------
// Weight transpose + fp32 -> bf16:  W[K][N]  ->  WT[N][K]
// ---------------------------------------------------------------------------
__global__ void k_weight_T(const float* __restrict__ W, int Krows, int Ncols,
                           u16* __restrict__ WT) {
  int idx = blockIdx.x * blockDim.x + threadIdx.x;
  if (idx >= Krows * Ncols) return;
  int k = idx / Ncols, n = idx % Ncols;
  WT[(size_t)n * Krows + k] = f2bf(W[idx]);
}

// ---------------------------------------------------------------------------
// LayerNorm over last dim (C = 64/128), one wave per row, bf16 output.
// tr==1 reads the transposed view: src row (i*L+n) for logical row m=(n,i).
// ---------------------------------------------------------------------------
__global__ void k_layernorm(const float* __restrict__ src, int C, int tr,
                            const float* __restrict__ g,
                            const float* __restrict__ b,
                            u16* __restrict__ dst) {
  const int lane = threadIdx.x & 31;
  const int wave = threadIdx.x >> 5;
  const int m = blockIdx.x * 8 + wave;          // (n,i)
  const int nn = m / L, ii = m % L;
  const size_t srow = (size_t)(tr ? (ii * L + nn) : m) * C;
  float xr[4];
  int cnt = 0;
  float s = 0.f, s2 = 0.f;
  for (int c = lane; c < C; c += 32) {
    float x = src[srow + c];
    xr[cnt++] = x;
    s += x; s2 += x * x;
  }
#pragma unroll
  for (int o = 16; o >= 1; o >>= 1) {
    s  += __shfl_xor(s, o, 32);
    s2 += __shfl_xor(s2, o, 32);
  }
  float mean = s / C;
  float var  = s2 / C - mean * mean;
  float rstd = rsqrtf(var + 1e-5f);
  cnt = 0;
  for (int c = lane; c < C; c += 32) {
    float y = (xr[cnt++] - mean) * rstd * g[c] + b[c];
    dst[(size_t)m * C + c] = f2bf(y);
  }
}

// ---------------------------------------------------------------------------
// Generic GEMM: D[m][n] = A[m][:] . Bt[n][:]  (A MxK bf16, Bt NxK bf16)
// Epilogue: optional bias for n>=bias_from, activation for n>=act_from
// (1=relu, 2=sigmoid). bf16 output. Wave computes one 16x16 tile.
// ---------------------------------------------------------------------------
__global__ void k_gemm_bias_act(const u16* __restrict__ A, int lda,
                                const u16* __restrict__ Bt, int ldb,
                                int Mrows, int Kdim,
                                const float* __restrict__ bias, int bias_from,
                                int act, int act_from,
                                u16* __restrict__ D, int ldd) {
  const int lane = threadIdx.x & 31;
  const int wave = threadIdx.x >> 5;
  const int m0 = (blockIdx.y * 8 + wave) * 16;
  if (m0 >= Mrows) return;
  const int n0 = blockIdx.x * 16;
  const u16* pa = A + (size_t)m0 * lda;
  const u16* pb = Bt + (size_t)n0 * ldb;
  v8f acc = {};
  for (int k0 = 0; k0 < Kdim; k0 += 32) {
    __builtin_prefetch(pa + k0 + 128, 0, 0);   // global_prefetch_b8
    acc = wmma_acc(load_frag(pa + k0, lda, lane),
                   load_frag(pb + k0, ldb, lane), acc);
  }
  const int n = n0 + (lane & 15);
#pragma unroll
  for (int r = 0; r < 8; ++r) {
    int m = m0 + r + ((lane >> 4) << 3);
    float x = acc[r];
    if (bias && n >= bias_from) x += bias[n - bias_from];
    if (n >= act_from) {
      if (act == 1)      x = fmaxf(x, 0.0f);
      else if (act == 2) x = 1.0f / (1.0f + __expf(-x));
    }
    D[(size_t)m * ldd + n] = f2bf(x);
  }
}

// ---------------------------------------------------------------------------
// GEMM + residual add, fp32 out (Ncols == DE).  transpose_out maps logical
// row m=(n,i) to destination row (i*L+n) (undo axial transpose).
// ---------------------------------------------------------------------------
__global__ void k_gemm_resid(const u16* __restrict__ A, int lda,
                             const u16* __restrict__ Bt, int ldb,
                             int Mrows, int Kdim,
                             const float* __restrict__ bias,
                             const float* __restrict__ resid,
                             float* __restrict__ dst,
                             int transpose_out) {
  const int lane = threadIdx.x & 31;
  const int wave = threadIdx.x >> 5;
  const int m0 = (blockIdx.y * 8 + wave) * 16;
  if (m0 >= Mrows) return;
  const int n0 = blockIdx.x * 16;
  const u16* pa = A + (size_t)m0 * lda;
  const u16* pb = Bt + (size_t)n0 * ldb;
  v8f acc = {};
  for (int k0 = 0; k0 < Kdim; k0 += 32) {
    __builtin_prefetch(pa + k0 + 128, 0, 0);
    acc = wmma_acc(load_frag(pa + k0, lda, lane),
                   load_frag(pb + k0, ldb, lane), acc);
  }
  const int n = n0 + (lane & 15);
#pragma unroll
  for (int r = 0; r < 8; ++r) {
    int m  = m0 + r + ((lane >> 4) << 3);
    int nn = m / L, ii = m % L;
    int drow = transpose_out ? (ii * L + nn) : m;
    float x = acc[r];
    if (bias) x += bias[n];
    dst[(size_t)drow * DE + n] = resid[(size_t)drow * DE + n] + x;
  }
}

// ---------------------------------------------------------------------------
// Attention logits with double-buffered async global->LDS panel staging:
//   out[h][i][j] = sum_{n,k} Qs[h][i][n*48+k] * Ks[h][j][n*48+k]
// Block = 256 threads = 8 waves. Block tile = 128 i-rows x 16 j-cols.
// Per 64-deep K-chunk: A panel 128x64 (4 x b128 / thread), B panel 16x64
// (1 x b64 / thread) -> uniform 5 async issues per wave per chunk, so
// s_wait_asynccnt 5 retires exactly the chunk being consumed while the next
// chunk's transfers remain in flight.
// ---------------------------------------------------------------------------
__global__ void k_gemm_logits_lds(const u16* __restrict__ Qs,
                                  const u16* __restrict__ Ks,
                                  float* __restrict__ out) {
  __shared__ u16 Apan[2][128 * 64];   // 2 x 16 KB
  __shared__ u16 Bpan[2][16 * 64];    // 2 x  2 KB
  const int tid  = threadIdx.x;
  const int lane = tid & 31;
  const int wave = tid >> 5;
  const int h  = blockIdx.z;
  const int i0 = blockIdx.y * 128;
  const int j0 = blockIdx.x * 16;
  const u16* Abase = Qs + ((size_t)h * L + i0) * KD;
  const u16* Bbase = Ks + ((size_t)h * L + j0) * KD;

  // thread's staging slice (fixed across chunks)
  const int ar = tid >> 1, ac = (tid & 1) * 32;   // A: row, 32-half slice
  const int br = tid >> 4, bc = (tid & 15) * 4;   // B: row, 4-half slice

  auto stage = [&](int chunk, int buf) {
    const int kc = chunk * 64;
    const u16* ag = Abase + (size_t)ar * KD + kc + ac;
    u16*       al = &Apan[buf][ar * 64 + ac];
    const u16* bg = Bbase + (size_t)br * KD + kc + bc;
    u16*       bl = &Bpan[buf][br * 64 + bc];
#ifdef HAVE_ASYNC_LDS
#pragma unroll
    for (int q = 0; q < 4; ++q)
      __builtin_amdgcn_global_load_async_to_lds_b128(
          (g_v4p)(ag + 8 * q), (l_v4p)(al + 8 * q), 0, 0);
    __builtin_amdgcn_global_load_async_to_lds_b64((g_v2p)bg, (l_v2p)bl, 0, 0);
#else
#pragma unroll
    for (int q = 0; q < 4; ++q)
      *(ushort8v*)(al + 8 * q) = *(const ushort8v*)(ag + 8 * q);
    *(uint2v*)bl = *(const uint2v*)bg;
#endif
  };

  constexpr int NC = KD / 64;  // 288 chunks
  stage(0, 0);
  stage(1, 1);
  v8f acc = {};
  for (int c = 0; c < NC; ++c) {
    if (c + 2 < NC) { WAIT_ASYNC(5); } else { WAIT_ASYNC(0); }
    __syncthreads();                 // chunk c fully staged by all waves
    const int buf = c & 1;
    const u16* ap = &Apan[buf][(wave * 16) * 64];
    const u16* bp = &Bpan[buf][0];
    acc = wmma_acc(load_frag(ap,      64, lane), load_frag(bp,      64, lane), acc);
    acc = wmma_acc(load_frag(ap + 32, 64, lane), load_frag(bp + 32, 64, lane), acc);
    __syncthreads();                 // all waves done reading buf
    if (c + 2 < NC) stage(c + 2, buf);
  }
  const int j = j0 + (lane & 15);
#pragma unroll
  for (int r = 0; r < 8; ++r) {
    int i = i0 + wave * 16 + r + ((lane >> 4) << 3);
    out[((size_t)h * L + i) * L + j] = acc[r];
  }
}

// ---------------------------------------------------------------------------
// attn x V with fused gate: O[(n,i)][h*48+d] =
//   gate[n,i,h,d] * sum_j attn[h][i][j] * Vt[h][n][d][j]
// blockIdx.z = h*L + n ; x tiles d (3) ; y*8+wave tiles i (24)
// ---------------------------------------------------------------------------
__global__ void k_gemm_attnout(const u16* __restrict__ attn,
                               const u16* __restrict__ Vt,
                               const u16* __restrict__ qkvg,  // gate @ col 576
                               u16* __restrict__ O) {
  const int lane = threadIdx.x & 31;
  const int wave = threadIdx.x >> 5;
  const int z = blockIdx.z;
  const int h = z / L, n = z % L;
  const int m0 = (blockIdx.y * 8 + wave) * 16;  // i tile
  const int n0 = blockIdx.x * 16;               // d tile
  const u16* A = attn + (size_t)h * L * L + (size_t)m0 * L;
  const u16* B = Vt + ((size_t)h * L + n) * DH * L + (size_t)n0 * L;
  v8f acc = {};
  for (int k0 = 0; k0 < L; k0 += 32)
    acc = wmma_acc(load_frag(A + k0, L, lane),
                   load_frag(B + k0, L, lane), acc);
  const int d = n0 + (lane & 15);
#pragma unroll
  for (int r = 0; r < 8; ++r) {
    int i = m0 + r + ((lane >> 4) << 3);
    size_t row = (size_t)n * L + i;
    float g = bf2f(qkvg[row * 768 + 576 + h * DH + d]);
    O[row * HD + h * DH + d] = f2bf(acc[r] * g);
  }
}

// ---------------------------------------------------------------------------
// RoPE + scale + layout shuffle:
// reads fused projection qkvg[(n,i)][seg*192 + h*48 + k] (q=seg0, k=seg1)
// writes Qs/Ks [h][i][n*48+k] bf16 (contraction index = n*48+k).
// ---------------------------------------------------------------------------
__global__ void k_rope(const u16* __restrict__ qkvg,
                       const int* __restrict__ pos,
                       u16* __restrict__ Qs, u16* __restrict__ Ks) {
  int idx = blockIdx.x * blockDim.x + threadIdx.x;
  if (idx >= L * L * NH * 24) return;
  int k = idx % 24;
  int h = (idx / 24) % NH;
  int i = (idx / (24 * NH)) % L;
  int n = idx / (24 * NH * L);
  size_t row = (size_t)n * L + i;
  const u16* qp = qkvg + row * 768 + 0   + h * DH;
  const u16* kp = qkvg + row * 768 + 192 + h * DH;
  float inv = __expf(-logf(10000.0f) * (float)k / 24.0f);
  float ang = (float)pos[i] * inv;
  float sn, cs;
  __sincosf(ang, &sn, &cs);
  float q1 = bf2f(qp[k]), q2 = bf2f(qp[k + 24]);
  float k1 = bf2f(kp[k]), k2 = bf2f(kp[k + 24]);
  const float qs = 0.14433756729740643f;  // 1/sqrt(48)
  const float ks = 1.0f / (float)L;
  size_t o = ((size_t)h * L + i) * KD + (size_t)n * DH;
  Qs[o + k]      = f2bf((q1 * cs - q2 * sn) * qs);
  Qs[o + k + 24] = f2bf((q1 * sn + q2 * cs) * qs);
  Ks[o + k]      = f2bf((k1 * cs - k2 * sn) * ks);
  Ks[o + k + 24] = f2bf((k1 * sn + k2 * cs) * ks);
}

// V rearrange: Vt[h][n][d][j] = qkvg[(n,j)][384 + h*48 + d]
__global__ void k_v_rearrange(const u16* __restrict__ qkvg,
                              u16* __restrict__ Vt) {
  int idx = blockIdx.x * blockDim.x + threadIdx.x;
  if (idx >= M * HD) return;
  int d = idx % DH;
  int h = (idx / DH) % NH;
  int j = (idx / HD) % L;
  int n = idx / (HD * L);
  Vt[(((size_t)h * L + n) * DH + d) * L + j] =
      qkvg[((size_t)n * L + j) * 768 + 384 + h * DH + d];
}

// ---------------------------------------------------------------------------
// biasH[h][i][j] = LN(edge_bias[tr ? (j,i) : (i,j)]) . Wb[:,h]
// one wave per (i,j): 64 channels -> 2 per lane, shfl reductions.
// ---------------------------------------------------------------------------
__global__ void k_bias_proj(const float* __restrict__ edge_bias, int tr,
                            const float* __restrict__ g,
                            const float* __restrict__ b,
                            const float* __restrict__ Wb,  // [64][4]
                            float* __restrict__ biasH) {
  const int lane = threadIdx.x & 31;
  const int wave = threadIdx.x >> 5;
  const int m = blockIdx.x * 8 + wave;  // (i,j)
  const int i = m / L, j = m % L;
  const size_t srow = (size_t)(tr ? (j * L + i) : m) * DB;
  float x0 = edge_bias[srow + lane];
  float x1 = edge_bias[srow + lane + 32];
  float s = x0 + x1, s2 = x0 * x0 + x1 * x1;
#pragma unroll
  for (int o = 16; o >= 1; o >>= 1) {
    s  += __shfl_xor(s, o, 32);
    s2 += __shfl_xor(s2, o, 32);
  }
  float mean = s / DB;
  float var  = s2 / DB - mean * mean;
  float rstd = rsqrtf(var + 1e-5f);
  float y0 = (x0 - mean) * rstd * g[lane] + b[lane];
  float y1 = (x1 - mean) * rstd * g[lane + 32] + b[lane + 32];
#pragma unroll
  for (int h = 0; h < NH; ++h) {
    float p = y0 * Wb[lane * NH + h] + y1 * Wb[(lane + 32) * NH + h];
#pragma unroll
    for (int o = 16; o >= 1; o >>= 1) p += __shfl_xor(p, o, 32);
    if (lane == 0) biasH[((size_t)h * L + i) * L + j] = p;
  }
}

// ---------------------------------------------------------------------------
// softmax over j (+bias before, +INF*(mask-1) after, per reference order).
// block = 128 threads, 3 j's each, LDS tree reductions.
// ---------------------------------------------------------------------------
__global__ void k_softmax_mask(const float* __restrict__ logits,
                               const float* __restrict__ biasH,
                               const float* __restrict__ mask,
                               u16* __restrict__ out) {
  __shared__ float red[128];
  const int hi = blockIdx.x;  // h*L + i
  const int i = hi % L;
  const int t = threadIdx.x;
  float v[3];
#pragma unroll
  for (int p = 0; p < 3; ++p) {
    int j = t + p * 128;
    v[p] = logits[(size_t)hi * L + j] + biasH[(size_t)hi * L + j];
  }
  float mx = fmaxf(fmaxf(v[0], v[1]), v[2]);
  red[t] = mx;
  __syncthreads();
  for (int o = 64; o >= 1; o >>= 1) {
    if (t < o) red[t] = fmaxf(red[t], red[t + o]);
    __syncthreads();
  }
  mx = red[0];
  __syncthreads();
  float e[3], s = 0.f;
#pragma unroll
  for (int p = 0; p < 3; ++p) { e[p] = __expf(v[p] - mx); s += e[p]; }
  red[t] = s;
  __syncthreads();
  for (int o = 64; o >= 1; o >>= 1) {
    if (t < o) red[t] += red[t + o];
    __syncthreads();
  }
  float invs = 1.0f / red[0];
#pragma unroll
  for (int p = 0; p < 3; ++p) {
    int j = t + p * 128;
    float a = e[p] * invs + INFV * (mask[(size_t)i * L + j] - 1.0f);
    out[(size_t)hi * L + j] = f2bf(a);
  }
}

// ---------------------------------------------------------------------------
// Host side
// ---------------------------------------------------------------------------
struct AttnP {
  const float *Wb, *Wg, *Wk, *Wo, *Wq, *Wv, *bg, *ln_b_b, *ln_b_g, *ln_e_b, *ln_e_g;
};

extern "C" void kernel_launch(void* const* d_in, const int* in_sizes, int n_in,
                              void* d_out, int out_size, void* d_ws, size_t ws_size,
                              hipStream_t stream) {
  (void)in_sizes; (void)n_in; (void)out_size; (void)ws_size;
  // d_in order: jax pytree flatten with dict keys sorted:
  // 0 edge, 1 edge_bias, 2 edge_pos, 3 edge_mask,
  // params: 'col' -> 4..14, 'ff' -> 15..20, 'row' -> 21..31
  // attn dict sorted: Wb,Wg,Wk,Wo,Wq,Wv,bg,ln_b_b,ln_b_g,ln_e_b,ln_e_g
  // ff dict sorted:   W1,W2,b1,b2,ln_b,ln_g
  const float* edge      = (const float*)d_in[0];
  const float* edge_bias = (const float*)d_in[1];
  const int*   edge_pos  = (const int*)d_in[2];
  const float* mask      = (const float*)d_in[3];
  auto getA = [&](int base) {
    AttnP p;
    p.Wb = (const float*)d_in[base + 0];  p.Wg = (const float*)d_in[base + 1];
    p.Wk = (const float*)d_in[base + 2];  p.Wo = (const float*)d_in[base + 3];
    p.Wq = (const float*)d_in[base + 4];  p.Wv = (const float*)d_in[base + 5];
    p.bg = (const float*)d_in[base + 6];
    p.ln_b_b = (const float*)d_in[base + 7];  p.ln_b_g = (const float*)d_in[base + 8];
    p.ln_e_b = (const float*)d_in[base + 9];  p.ln_e_g = (const float*)d_in[base + 10];
    return p;
  };
  AttnP colP = getA(4);
  AttnP rowP = getA(21);
  const float* W1   = (const float*)d_in[15];
  const float* W2   = (const float*)d_in[16];
  const float* b1   = (const float*)d_in[17];
  const float* b2   = (const float*)d_in[18];
  const float* fflb = (const float*)d_in[19];
  const float* fflg = (const float*)d_in[20];
  float* outp = (float*)d_out;

  // scratch arena
  char* ws = (char*)d_ws;
  size_t off = 0;
  auto alloc = [&](size_t bytes) -> void* {
    void* p = ws + off;
    off += (bytes + 255) & ~(size_t)255;
    return p;
  };
  u16* eL    = (u16*)alloc((size_t)M * DE * 2);       //  37.7 MB
  u16* qkvg  = (u16*)alloc((size_t)M * 768 * 2);      // 226.5 MB (FF hidden aliases this)
  u16* Qs    = (u16*)alloc((size_t)NH * L * KD * 2);  //  56.6 MB
  u16* Ks    = (u16*)alloc((size_t)NH * L * KD * 2);  //  56.6 MB
  u16* Vt    = (u16*)alloc((size_t)NH * L * KD * 2);  //  56.6 MB
  u16* Obuf  = (u16*)alloc((size_t)M * HD * 2);       //  56.6 MB
  float* biasH  = (float*)alloc((size_t)NH * L * L * 4);
  float* logits = (float*)alloc((size_t)NH * L * L * 4);
  u16*   attnbf = (u16*)alloc((size_t)NH * L * L * 2);
  u16* WT   = (u16*)alloc((size_t)768 * DE * 2);      // fused q|k|v|g weights, NxK
  u16* WoT  = (u16*)alloc((size_t)DE * HD * 2);
  u16* W1T  = (u16*)alloc((size_t)256 * DE * 2);
  u16* W2T  = (u16*)alloc((size_t)DE * 256 * 2);
  u16* Hbuf = qkvg;  // alias: FF hidden (M x 256 bf16) reuses qkvg region

  const int TPB = 256;
  auto cdiv = [](long a, long b) { return (int)((a + b - 1) / b); };

  auto axial = [&](const AttnP& p, const float* e_in, const float* resid,
                   float* e_out, int tr) {
    // weights -> bf16 NxK (q,k,v,g fused into 768x128; Wo -> 128x192)
    int wcnt = DE * HD;
    k_weight_T<<<cdiv(wcnt, TPB), TPB, 0, stream>>>(p.Wq, DE, HD, WT + 0 * HD * DE);
    k_weight_T<<<cdiv(wcnt, TPB), TPB, 0, stream>>>(p.Wk, DE, HD, WT + 1 * HD * DE);
    k_weight_T<<<cdiv(wcnt, TPB), TPB, 0, stream>>>(p.Wv, DE, HD, WT + 2 * HD * DE);
    k_weight_T<<<cdiv(wcnt, TPB), TPB, 0, stream>>>(p.Wg, DE, HD, WT + 3 * HD * DE);
    k_weight_T<<<cdiv(HD * DE, TPB), TPB, 0, stream>>>(p.Wo, HD, DE, WoT);
    // LN(edge view)
    k_layernorm<<<M / 8, TPB, 0, stream>>>(e_in, DE, tr, p.ln_e_g, p.ln_e_b, eL);
    // fused q|k|v|gate projection (sigmoid+bias on gate segment cols 576..767)
    k_gemm_bias_act<<<dim3(768 / 16, (M / 16) / 8), TPB, 0, stream>>>(
        eL, DE, WT, DE, M, DE, p.bg, 576, 2, 576, qkvg, 768);
    // RoPE + scaling + layout
    k_rope<<<cdiv((long)L * L * NH * 24, TPB), TPB, 0, stream>>>(qkvg, edge_pos, Qs, Ks);
    k_v_rearrange<<<cdiv((long)M * HD, TPB), TPB, 0, stream>>>(qkvg, Vt);
    // bias head projection
    k_bias_proj<<<M / 8, TPB, 0, stream>>>(edge_bias, tr, p.ln_b_g, p.ln_b_b, p.Wb, biasH);
    // logits (async-LDS staged), softmax, attn x V (gated)
    k_gemm_logits_lds<<<dim3(L / 16, L / 128, NH), TPB, 0, stream>>>(Qs, Ks, logits);
    k_softmax_mask<<<NH * L, 128, 0, stream>>>(logits, biasH, mask, attnbf);
    k_gemm_attnout<<<dim3(DH / 16, (L / 16) / 8, NH * L), TPB, 0, stream>>>(
        attnbf, Vt, qkvg, Obuf);
    // output projection + residual (+ transpose-back for the row pass)
    k_gemm_resid<<<dim3(DE / 16, (M / 16) / 8), TPB, 0, stream>>>(
        Obuf, HD, WoT, HD, M, HD, nullptr, resid, e_out, tr);
  };

  // row pass: transposed view, residual = input edge, writes d_out
  axial(rowP, edge, edge, outp, 1);
  // col pass: natural view, in-place residual on d_out
  axial(colP, outp, outp, outp, 0);

  // feed-forward: LN -> (128->256, +b1, relu) -> (256->128, +b2) + residual
  k_layernorm<<<M / 8, TPB, 0, stream>>>(outp, DE, 0, fflg, fflb, eL);
  k_weight_T<<<cdiv(DE * 256, TPB), TPB, 0, stream>>>(W1, DE, 256, W1T);
  k_weight_T<<<cdiv(256 * DE, TPB), TPB, 0, stream>>>(W2, 256, DE, W2T);
  k_gemm_bias_act<<<dim3(256 / 16, (M / 16) / 8), TPB, 0, stream>>>(
      eL, DE, W1T, DE, M, DE, b1, 0, 1, 0, Hbuf, 256);
  k_gemm_resid<<<dim3(DE / 16, (M / 16) / 8), TPB, 0, stream>>>(
      Hbuf, 256, W2T, 256, M, 256, b2, outp, outp, 0);
}
